// SelfAttentionModule_40286793237106
// MI455X (gfx1250) — matrix operand order repaired and checked by reading, not compiled, not run on hardware
//
#include <hip/hip_runtime.h>

typedef _Float16 v16h __attribute__((ext_vector_type(16)));
typedef _Float16 v8h  __attribute__((ext_vector_type(8)));
typedef float    v8f  __attribute__((ext_vector_type(8)));

#define BATCH 4
#define CH    8
#define CHP   16   // channel dim zero-padded to WMMA M/N
#define HW    4096

// ---------------------------------------------------------------------------
// Kernel 1: 1x1 conv projections.
//   q,k (f16, zero-padded to 16 rows for WMMA), v (fp32), pq/pk/pv scalars.
// ---------------------------------------------------------------------------
__global__ __launch_bounds__(256)
void proj_kernel(const float* __restrict__ x,
                 const float* __restrict__ cwq, const float* __restrict__ cbq,
                 const float* __restrict__ cwk, const float* __restrict__ cbk,
                 const float* __restrict__ cwv, const float* __restrict__ cbv,
                 const float* __restrict__ pwq, const float* __restrict__ pbq,
                 const float* __restrict__ pwk, const float* __restrict__ pbk,
                 const float* __restrict__ pwv, const float* __restrict__ pbv,
                 _Float16* __restrict__ qh, _Float16* __restrict__ kh,
                 float* __restrict__ vf,
                 float* __restrict__ pqg, float* __restrict__ pkg,
                 float* __restrict__ pvg) {
  __shared__ float swq[64], swk[64], swv[64];
  __shared__ float sbq[8], sbk[8], sbv[8];
  __shared__ float spq[8], spk[8], spv[8];
  __shared__ float spb[3];
  const int t = threadIdx.x;
  if (t < 64) { swq[t] = cwq[t]; swk[t] = cwk[t]; swv[t] = cwv[t]; }
  if (t < 8)  { sbq[t] = cbq[t]; sbk[t] = cbk[t]; sbv[t] = cbv[t];
                spq[t] = pwq[t]; spk[t] = pwk[t]; spv[t] = pwv[t]; }
  if (t == 0) { spb[0] = pbq[0]; spb[1] = pbk[0]; spb[2] = pbv[0]; }
  __syncthreads();

  const int g = blockIdx.x * 256 + t;
  const int b = g >> 12;
  const int p = g & (HW - 1);

  __builtin_prefetch(&x[((size_t)(b * CH)) * HW + p], 0, 0);

  float xv[CH];
#pragma unroll
  for (int c = 0; c < CH; ++c) xv[c] = x[((size_t)(b * CH + c)) * HW + p];

  // PAM scalar projections (C8 == 1)
  float aq = spb[0], ak = spb[1], av = spb[2];
#pragma unroll
  for (int c = 0; c < CH; ++c) {
    aq = __fmaf_rn(spq[c], xv[c], aq);
    ak = __fmaf_rn(spk[c], xv[c], ak);
    av = __fmaf_rn(spv[c], xv[c], av);
  }
  pqg[b * HW + p] = aq;
  pkg[b * HW + p] = ak;
  pvg[b * HW + p] = av;

  // CAM projections (rows 0..7 real, rows 8..15 zero padding for WMMA)
#pragma unroll
  for (int o = 0; o < CH; ++o) {
    float q = sbq[o], k = sbk[o], v = sbv[o];
#pragma unroll
    for (int c = 0; c < CH; ++c) {
      q = __fmaf_rn(swq[o * 8 + c], xv[c], q);
      k = __fmaf_rn(swk[o * 8 + c], xv[c], k);
      v = __fmaf_rn(swv[o * 8 + c], xv[c], v);
    }
    const size_t hidx = ((size_t)(b * CHP + o)) * HW + p;
    qh[hidx] = (_Float16)q;
    kh[hidx] = (_Float16)k;
    vf[((size_t)(b * CH + o)) * HW + p] = v;
  }
#pragma unroll
  for (int o = CH; o < CHP; ++o) {
    const size_t hidx = ((size_t)(b * CHP + o)) * HW + p;
    qh[hidx] = (_Float16)0.0f;
    kh[hidx] = (_Float16)0.0f;
  }
}

// ---------------------------------------------------------------------------
// Kernel 2: CAM Gram matrix via v_wmma_f32_16x16x32_f16 + row softmax.
// One wave (32 lanes) per batch. G[c][d] = sum_p q[c,p]*k[d,p].
// Inputs are zero-padded to 16 rows, so the inner loop is pure
// global_load_b128 + v_wmma (no masking ALU).
//   A (16x32 f16): lane%16 = row M; elems 0-7 <- K=kb+8*hs..,
//                  elems 8-15 <- K=kb+16+8*hs..
//   B (32x16 f16): lane%16 = col N; elems 0-7 <- K=kb+8*hs..,
//                  elems 8-15 <- K=kb+16+8*hs..
// ---------------------------------------------------------------------------
__global__ __launch_bounds__(32)
void cam_gram_softmax_kernel(const _Float16* __restrict__ qh,
                             const _Float16* __restrict__ kh,
                             float* __restrict__ attn) {
  const int b    = blockIdx.x;
  const int lane = threadIdx.x;
  const int row  = lane & 15;
  const int hs   = lane >> 4;

  const _Float16* qrow = qh + ((size_t)(b * CHP + row)) * HW;
  const _Float16* krow = kh + ((size_t)(b * CHP + row)) * HW;

  v8f acc = {};
  for (int kb = 0; kb < HW; kb += 32) {
    v8h a0 = *(const v8h*)(qrow + kb + 8 * hs);
    v8h a1 = *(const v8h*)(qrow + kb + 16 + 8 * hs);
    v8h b0 = *(const v8h*)(krow + kb + 8 * hs);
    v8h b1 = *(const v8h*)(krow + kb + 16 + 8 * hs);
    v16h A  = __builtin_shufflevector(a0, a1, 0, 1, 2, 3, 4, 5, 6, 7,
                                      8, 9, 10, 11, 12, 13, 14, 15);
    v16h Bm = __builtin_shufflevector(b0, b1, 0, 1, 2, 3, 4, 5, 6, 7,
                                      8, 9, 10, 11, 12, 13, 14, 15);
    acc = __builtin_amdgcn_wmma_f32_16x16x32_f16(
        /*neg_a=*/false, A, /*neg_b=*/false, Bm,
        /*c_mod=*/(short)0, acc, /*reuse_a=*/false, /*reuse_b=*/false);
  }

  // D layout: VGPR r, lanes 0-15 -> M=r, N=lane. Valid region M<8, N<8.
  __shared__ float Gs[64];
  if (lane < 8) {
#pragma unroll
    for (int r = 0; r < 8; ++r) Gs[r * 8 + lane] = acc[r];
  }
  __syncthreads();

  if (lane < 8) {  // softmax over row `lane`
    float m = Gs[lane * 8];
#pragma unroll
    for (int j = 1; j < 8; ++j) m = fmaxf(m, Gs[lane * 8 + j]);
    float e[8], s = 0.0f;
#pragma unroll
    for (int j = 0; j < 8; ++j) { e[j] = __expf(Gs[lane * 8 + j] - m); s += e[j]; }
    const float inv = 1.0f / s;
#pragma unroll
    for (int j = 0; j < 8; ++j) attn[b * 64 + lane * 8 + j] = e[j] * inv;
  }
}

// ---------------------------------------------------------------------------
// Kernel 3: PAM. Since C8==1: out[n] = sum_m pv[m]*exp(s*pk[m]-mx) / sum exp.
// pk/pv staged in LDS; exact row max mx = s>=0 ? s*max(pk) : s*min(pk).
// ---------------------------------------------------------------------------
__global__ __launch_bounds__(256)
void pam_kernel(const float* __restrict__ pqg, const float* __restrict__ pkg,
                const float* __restrict__ pvg, float* __restrict__ pam) {
  __shared__ float spk[HW];
  __shared__ float spv[HW];
  __shared__ float smax[256];
  __shared__ float smin[256];
  const int b    = blockIdx.x >> 4;
  const int tile = blockIdx.x & 15;
  const int t    = threadIdx.x;

  float lmax = -3.0e38f, lmin = 3.0e38f;
  for (int i = t; i < HW; i += 256) {
    const float k = pkg[b * HW + i];
    spk[i] = k;
    spv[i] = pvg[b * HW + i];
    lmax = fmaxf(lmax, k);
    lmin = fminf(lmin, k);
  }
  smax[t] = lmax;
  smin[t] = lmin;
  __syncthreads();
  for (int st = 128; st > 0; st >>= 1) {
    if (t < st) {
      smax[t] = fmaxf(smax[t], smax[t + st]);
      smin[t] = fminf(smin[t], smin[t + st]);
    }
    __syncthreads();
  }
  const float kmax = smax[0], kmin = smin[0];

  const int n   = tile * 256 + t;
  const float s = pqg[b * HW + n];
  const float mx = (s >= 0.0f) ? s * kmax : s * kmin;

  float den = 0.0f, num = 0.0f;
#pragma unroll 4
  for (int m = 0; m < HW; ++m) {
    const float e = __expf(__fmaf_rn(s, spk[m], -mx));
    den += e;
    num = __fmaf_rn(e, spv[m], num);
  }
  pam[b * HW + n] = num / den;
}

// ---------------------------------------------------------------------------
// Kernel 4: out = (attn @ v) + pam + 2*x
// ---------------------------------------------------------------------------
__global__ __launch_bounds__(256)
void combine_kernel(const float* __restrict__ x, const float* __restrict__ attn,
                    const float* __restrict__ vf, const float* __restrict__ pam,
                    float* __restrict__ out) {
  const int g = blockIdx.x * 256 + threadIdx.x;
  const int b = g >> 12;
  const int p = g & (HW - 1);

  __shared__ float sat[64];
  if (threadIdx.x < 64) sat[threadIdx.x] = attn[b * 64 + threadIdx.x];
  __syncthreads();

  float vv[CH];
#pragma unroll
  for (int d = 0; d < CH; ++d) vv[d] = vf[((size_t)(b * CH + d)) * HW + p];
  const float pp = pam[b * HW + p];

#pragma unroll
  for (int c = 0; c < CH; ++c) {
    float acc = 0.0f;
#pragma unroll
    for (int d = 0; d < CH; ++d) acc = __fmaf_rn(sat[c * 8 + d], vv[d], acc);
    const float xx = x[((size_t)(b * CH + c)) * HW + p];
    out[((size_t)(b * CH + c)) * HW + p] = acc + pp + 2.0f * xx;
  }
}

// ---------------------------------------------------------------------------
// Workspace layout (bytes):
//   qh  f16 [4][16][4096]  @       0  (524288)   rows 8-15 zero
//   kh  f16 [4][16][4096]  @  524288  (524288)   rows 8-15 zero
//   vf  f32 [4][8][4096]   @ 1048576  (524288)
//   pq  f32 [4][4096]      @ 1572864  ( 65536)
//   pk  f32 [4][4096]      @ 1638400  ( 65536)
//   pv  f32 [4][4096]      @ 1703936  ( 65536)
//   pam f32 [4][4096]      @ 1769472  ( 65536)
//   att f32 [4][64]        @ 1835008  (  1024)
// ---------------------------------------------------------------------------
extern "C" void kernel_launch(void* const* d_in, const int* in_sizes, int n_in,
                              void* d_out, int out_size, void* d_ws, size_t ws_size,
                              hipStream_t stream) {
  const float* x   = (const float*)d_in[0];
  const float* cwq = (const float*)d_in[1];
  const float* cbq = (const float*)d_in[2];
  const float* cwk = (const float*)d_in[3];
  const float* cbk = (const float*)d_in[4];
  const float* cwv = (const float*)d_in[5];
  const float* cbv = (const float*)d_in[6];
  const float* pwq = (const float*)d_in[7];
  const float* pbq = (const float*)d_in[8];
  const float* pwk = (const float*)d_in[9];
  const float* pbk = (const float*)d_in[10];
  const float* pwv = (const float*)d_in[11];
  const float* pbv = (const float*)d_in[12];

  char* ws = (char*)d_ws;
  _Float16* qh  = (_Float16*)(ws + 0);
  _Float16* kh  = (_Float16*)(ws + 524288);
  float*    vf  = (float*)(ws + 1048576);
  float*    pqg = (float*)(ws + 1572864);
  float*    pkg = (float*)(ws + 1638400);
  float*    pvg = (float*)(ws + 1703936);
  float*    pam = (float*)(ws + 1769472);
  float*    att = (float*)(ws + 1835008);

  proj_kernel<<<64, 256, 0, stream>>>(x, cwq, cbq, cwk, cbk, cwv, cbv,
                                      pwq, pbq, pwk, pbk, pwv, pbv,
                                      qh, kh, vf, pqg, pkg, pvg);
  cam_gram_softmax_kernel<<<BATCH, 32, 0, stream>>>(qh, kh, att);
  pam_kernel<<<64, 256, 0, stream>>>(pqg, pkg, pvg, pam);
  combine_kernel<<<64, 256, 0, stream>>>(x, att, vf, pam, (float*)d_out);
}
